// LongAttention_30090540875957
// MI455X (gfx1250) — compile-verified, matrix operand order checked by reading, not run on hardware
//
#include <hip/hip_runtime.h>
#include <hip/hip_bf16.h>
#include <math.h>
#include <stdint.h>

typedef __attribute__((ext_vector_type(16))) _Float16 v16h;
typedef __attribute__((ext_vector_type(8)))  float    v8f;

// ---------------------------------------------------------------------------
// problem constants
// ---------------------------------------------------------------------------
#define BQ    2
#define OBJ   4
#define TT    8
#define HW    1024          // 32*32
#define THW   8192          // 8*1024
#define DMODEL 256
#define DQK    512
#define HEADS  8
#define DQ     64
#define DV     32
#define CIN    256          // all three convs have Cin=256
#define PHW    1156         // 34*34 padded plane
#define KD9    (9 * CIN)    // 2304 = implicit-GEMM K

// ---------------------------------------------------------------------------
// WMMA helpers (CDNA5 16x16x32 f16 -> f32)
// ---------------------------------------------------------------------------
__device__ __forceinline__ v8f wmma16(v16h a, v16h b, v8f c) {
  return __builtin_amdgcn_wmma_f32_16x16x32_f16(
      false, a, false, b, (short)0, c, false, false);
}

// 16x32 fragment of a row-major (rows x K) f16 matrix; lane = row%16,
// per-lane K pattern e<8 -> kb+e, e>=8 -> kb+16+(e-8), kb=(lane>>4)*8.
// Serves the A operand AND any B operand stored N-major (N x K row-major):
// both map lane&15 to the 16-wide M/N index. Two contiguous 16B loads / lane.
__device__ __forceinline__ v16h load_frag_a(const _Float16* p, int ld,
                                            int row0, int k0) {
  const int lane = threadIdx.x & 31;
  const int m    = lane & 15;
  const int kb   = (lane >> 4) * 8;
  const _Float16* r = p + (size_t)(row0 + m) * ld + k0 + kb;
  v16h f;
#pragma unroll
  for (int e = 0; e < 8; ++e) f[e] = r[e];
#pragma unroll
  for (int e = 0; e < 8; ++e) f[8 + e] = r[16 + e];
  return f;
}

// ---------------------------------------------------------------------------
// conversion / packing kernels
// ---------------------------------------------------------------------------
__global__ void cvt_f32_f16(const float* __restrict__ s, _Float16* __restrict__ d, int n) {
  int i = blockIdx.x * blockDim.x + threadIdx.x;
  if (i < n) d[i] = (_Float16)s[i];
}

// key (B,256,T,32,32) f32 -> (B*T,256,1024) f16
__global__ void pack_key_f16(const float* __restrict__ s, _Float16* __restrict__ d) {
  int i = blockIdx.x * blockDim.x + threadIdx.x;          // 2^22
  if (i >= BQ * TT * DMODEL * HW) return;
  int p = i & 1023, c = (i >> 10) & 255, t = (i >> 18) & 7, b = i >> 21;
  d[i] = (_Float16)s[(((size_t)(b * DMODEL + c) * TT + t) << 10) + p];
}

// value (B,OBJ,256,T,32,32) f32 -> (B*OBJ*T,256,1024) f16
__global__ void pack_value_f16(const float* __restrict__ s, _Float16* __restrict__ d) {
  int i = blockIdx.x * blockDim.x + threadIdx.x;          // 2^24
  if (i >= BQ * OBJ * TT * DMODEL * HW) return;
  int p = i & 1023, c = (i >> 10) & 255, t = (i >> 18) & 7;
  int o = (i >> 21) & 3, b = i >> 23;
  d[i] = (_Float16)s[((((size_t)((b * OBJ + o) * DMODEL + c)) * TT + t) << 10) + p];
}

// (NB, C=256, 32,32) f16 -> zero-padded NHWC (NB, 34*34, 256) f16.
// NHWC makes the im2col operand N-major: contiguous per-lane b128 loads.
__global__ void pad_hwc_f16(const _Float16* __restrict__ s, _Float16* __restrict__ d,
                            int NB) {
  int i = blockIdx.x * blockDim.x + threadIdx.x;
  if (i >= NB * PHW * CIN) return;
  int c   = i & 255;
  int t   = i >> 8;
  int nb  = t / PHW;
  int pos = t - nb * PHW;
  int yy = pos / 34, xx = pos - yy * 34;
  _Float16 v = (_Float16)0.f;
  if (yy >= 1 && yy <= 32 && xx >= 1 && xx <= 32)
    v = s[((size_t)(nb * CIN + c) << 10) + ((yy - 1) << 5) + (xx - 1)];
  d[i] = v;
}

// conv weight (Cout,Cin,3,3) f32 -> (Cout, 9, Cin) f16  (K' = rs*Cin + cin:
// every 32-wide K chunk stays inside one (rr,ss) block)
__global__ void repack_w_f16(const float* __restrict__ s, _Float16* __restrict__ d, int Cout) {
  int i = blockIdx.x * blockDim.x + threadIdx.x;
  if (i >= Cout * 9 * CIN) return;
  int cin = i & 255;
  int t   = i >> 8;
  int co  = t / 9;
  int rs  = t - co * 9;
  d[i] = (_Float16)s[((size_t)(co * CIN + cin)) * 9 + rs];
}

// ---------------------------------------------------------------------------
// 3x3 SAME conv as implicit GEMM:
//   Y(Cout, NB*HW) = Wr(Cout, 9*Cin) * P(pixel, 9*Cin)^T
// A (weights) double-buffered into LDS via global_load_async_to_lds_b128,
// B (NHWC patches) two contiguous b128 loads per lane.
// ---------------------------------------------------------------------------
__global__ void __launch_bounds__(32)
conv3x3_wmma(const _Float16* __restrict__ xhwc, // (NB, 34*34, Cin) f16
             const _Float16* __restrict__ wr,   // (Cout, 9*Cin)    f16
             const float*    __restrict__ bias,
             float*          __restrict__ y,    // (NB, Cout, 1024) f32
             int Cout) {
  const int cout0 = blockIdx.x * 16;
  const int n0    = blockIdx.y * 16;
  const int batch = n0 >> 10;
  const int pix0  = n0 & 1023;
  const int y0    = pix0 >> 5;
  const int xbase = pix0 & 31;
  const int lane  = threadIdx.x & 31;
  const _Float16* xb = xhwc + (size_t)batch * PHW * CIN;

  __shared__ __align__(16) _Float16 wtile[2][16 * 32];  // 2 x 1KB A tiles
  const int arow = lane & 15;
  const int aseg = lane >> 4;

  // async-stage one 16x32 f16 weight tile into LDS (2 x b128 per lane)
  auto issue = [&](int ci, int bufi) {
    uint64_t ga = (uint64_t)(size_t)(wr + (size_t)(cout0 + arow) * KD9 + ci * 32)
                + (uint64_t)(aseg * 16);
    unsigned la = (unsigned)(size_t)(&wtile[bufi][arow * 32]) + aseg * 16;
    asm volatile("global_load_async_to_lds_b128 %0, %1, off"
                 :: "v"(la), "v"(ga) : "memory");
    asm volatile("global_load_async_to_lds_b128 %0, %1, off"
                 :: "v"(la + 32u), "v"(ga + 32ull) : "memory");
  };

  const int NCH = KD9 / 32;                             // 72 chunks
  issue(0, 0);
  v8f acc = {};
  int ci = 0;
#pragma unroll
  for (int rr = 0; rr < 3; ++rr) {
#pragma unroll
    for (int ss = 0; ss < 3; ++ss) {
      const int prow = (y0 + rr) * 34 + xbase + ss;     // im2col pixel-row base
      __builtin_prefetch(xb + (size_t)(prow + 34) * CIN, 0, 1);
      for (int c8 = 0; c8 < 8; ++c8, ++ci) {
        v16h bf = load_frag_a(xb, CIN, prow, c8 * 32);  // NHWC patch fragment
        if (ci + 1 < NCH) {
          issue(ci + 1, (ci + 1) & 1);
          asm volatile("s_wait_asynccnt 0x2" ::: "memory");  // chunk ci landed
        } else {
          asm volatile("s_wait_asynccnt 0x0" ::: "memory");
        }
        v16h a = load_frag_a(wtile[ci & 1], 32, 0, 0);
        acc = wmma16(a, bf, acc);
      }
    }
  }

  const int mh = (lane >> 4) * 8;
  const int nL = lane & 15;
#pragma unroll
  for (int v = 0; v < 8; ++v) {
    int co = cout0 + mh + v;
    y[((size_t)batch * Cout + co) * HW + pix0 + nL] = acc[v] + bias[co];
  }
}

// ---------------------------------------------------------------------------
// BatchNorm statistics (biased var over NB*HW per channel) -> scale/shift
// ---------------------------------------------------------------------------
__global__ void bn_stats(const float* __restrict__ y, int NB, int C,
                         const float* __restrict__ g, const float* __restrict__ bt,
                         float* __restrict__ scale, float* __restrict__ shift) {
  const int c = blockIdx.x, tid = threadIdx.x;
  const int total = NB * HW;
  float s = 0.f, s2 = 0.f;
  for (int i = tid; i < total; i += blockDim.x) {
    int nb = i >> 10, p = i & 1023;
    float v = y[((size_t)nb * C + c) * HW + p];
    s += v; s2 += v * v;
  }
  __shared__ float sh[256], sh2[256];
  sh[tid] = s; sh2[tid] = s2;
  __syncthreads();
  for (int st = 128; st > 0; st >>= 1) {
    if (tid < st) { sh[tid] += sh[tid + st]; sh2[tid] += sh2[tid + st]; }
    __syncthreads();
  }
  if (tid == 0) {
    float mean = sh[0] / total;
    float var  = sh2[0] / total - mean * mean;
    float sc   = g[c] * rsqrtf(var + 1e-5f);
    scale[c] = sc;
    shift[c] = bt[c] - mean * sc;
  }
}

// ---------------------------------------------------------------------------
// normalize + repack into WMMA-operand-friendly attention layouts
// ---------------------------------------------------------------------------
// convq (B,512,1024) -> q_att (B,8,1024,64)      [A operand: row=n, K=d]
__global__ void pack_q_att(const float* __restrict__ s, const float* __restrict__ sc,
                           const float* __restrict__ sh, _Float16* __restrict__ d) {
  int i = blockIdx.x * blockDim.x + threadIdx.x;          // 2^20
  if (i >= BQ * DQK * HW) return;
  int dd = i & 63, n = (i >> 6) & 1023, h = (i >> 16) & 7, b = i >> 19;
  int ch = h * DQ + dd;
  float v = s[(((size_t)b * DQK + ch) << 10) + n];
  d[i] = (_Float16)(v * sc[ch] + sh[ch]);
}

// convk (B*T,512,1024) -> k_attT (B,8,8192,64)   [B operand, N-major: row=m, K=d]
__global__ void pack_k_attT(const float* __restrict__ s, const float* __restrict__ sc,
                            const float* __restrict__ sh, _Float16* __restrict__ d) {
  int i = blockIdx.x * blockDim.x + threadIdx.x;          // 2^23
  if (i >= BQ * HEADS * THW * DQ) return;
  int dd = i & 63, m = (i >> 6) & 8191, h = (i >> 19) & 7, b = i >> 22;
  int n = m & 1023, t = m >> 10;
  int ch = h * DQ + dd, nb = b * TT + t;
  float v = s[(((size_t)nb * DQK + ch) << 10) + n];
  d[i] = (_Float16)(v * sc[ch] + sh[ch]);
}

// convv (B*OBJ*T,256,1024) -> v_attT (B,OBJ,8,32,8192) [B operand, N-major: row=dv, K=m]
__global__ void pack_v_attT(const float* __restrict__ s, const float* __restrict__ sc,
                            const float* __restrict__ sh, _Float16* __restrict__ d) {
  int i = blockIdx.x * blockDim.x + threadIdx.x;          // 2^24
  if (i >= BQ * OBJ * HEADS * DV * THW) return;
  int m = i & 8191, dd = (i >> 13) & 31, h = (i >> 18) & 7;
  int o = (i >> 21) & 3, b = i >> 23;
  int n = m & 1023, t = m >> 10;
  int ch = h * DV + dd, nb = (b * OBJ + o) * TT + t;
  float v = s[(((size_t)nb * DMODEL + ch) << 10) + n];
  d[i] = (_Float16)(v * sc[ch] + sh[ch]);
}

// ---------------------------------------------------------------------------
// fused attention: per wave = one (b,h, 16-row) query tile.
// Pass 1: online softmax (running max + rescaled running sum, merged
//         cross-lane with shfl_xor).  Pass 2: P*V with P recomputed by WMMA.
// Scores never touch global memory.
// ---------------------------------------------------------------------------
__global__ void __launch_bounds__(32)
attention_wmma(const _Float16* __restrict__ q_att,   // (B,8,1024,64)
               const _Float16* __restrict__ k_attT,  // (B,8,8192,64)
               const _Float16* __restrict__ v_attT,  // (B,4,8,32,8192)
               float* __restrict__ attn_out) {       // (1024, B*4, 256)
  const int lane = threadIdx.x & 31;
  const int nL   = lane & 15;
  const int mh   = (lane >> 4) * 8;

  const int wid  = blockIdx.x;
  const int rt   = wid & 63;
  const int bh   = wid >> 6;
  const int b    = bh >> 3;
  const int h    = bh & 7;
  const int row0 = rt * 16;

  const _Float16* Q  = q_att  + (size_t)(b * HEADS + h) * HW * DQ;
  const _Float16* KT = k_attT + (size_t)(b * HEADS + h) * THW * DQ;
  const float     sc = 0.125f;   // 1/sqrt(64)

  const v16h aq0 = load_frag_a(Q, DQ, row0, 0);
  const v16h aq1 = load_frag_a(Q, DQ, row0, 32);

  // ---- pass 1: online row max + rescaled row sum --------------------------
  float rm[8], rs[8];
#pragma unroll
  for (int v = 0; v < 8; ++v) { rm[v] = -3.0e38f; rs[v] = 0.f; }
  for (int mt = 0; mt < THW / 16; ++mt) {
    v16h bk0 = load_frag_a(KT, DQ, mt * 16, 0);
    v16h bk1 = load_frag_a(KT, DQ, mt * 16, 32);
    v8f c = {};
    c = wmma16(aq0, bk0, c);
    c = wmma16(aq1, bk1, c);
#pragma unroll
    for (int v = 0; v < 8; ++v) {
      float s  = c[v] * sc;
      float mn = fmaxf(rm[v], s);
      rs[v] = rs[v] * __expf(rm[v] - mn) + __expf(s - mn);
      rm[v] = mn;
    }
  }
  // cross-lane merge of (max,sum) within each 16-lane half (one matrix row)
  float pm[8], rinv[8];
#pragma unroll
  for (int v = 0; v < 8; ++v) {
    float m = rm[v], s = rs[v];
#pragma unroll
    for (int msk = 1; msk < 16; msk <<= 1) {
      float om = __shfl_xor(m, msk, 32);
      float os = __shfl_xor(s, msk, 32);
      float mn = fmaxf(m, om);
      s = s * __expf(m - mn) + os * __expf(om - mn);
      m = mn;
    }
    pm[v] = m;
    rinv[v] = 1.0f / s;
  }

  // ---- pass 2: P * V ------------------------------------------------------
  __shared__ __align__(16) _Float16 pbuf[16 * 32];
  v8f zero = {};
  v8f acc[OBJ][2];
#pragma unroll
  for (int o = 0; o < OBJ; ++o) { acc[o][0] = zero; acc[o][1] = zero; }

  for (int mt = 0; mt < THW / 32; ++mt) {
    const int col0 = mt * 32;
#pragma unroll
    for (int hc = 0; hc < 2; ++hc) {
      v16h bk0 = load_frag_a(KT, DQ, col0 + hc * 16, 0);
      v16h bk1 = load_frag_a(KT, DQ, col0 + hc * 16, 32);
      v8f c = {};
      c = wmma16(aq0, bk0, c);
      c = wmma16(aq1, bk1, c);
#pragma unroll
      for (int v = 0; v < 8; ++v) {
        float pr = __expf(c[v] * sc - pm[v]);
        pbuf[(mh + v) * 32 + hc * 16 + nL] = (_Float16)pr;  // C-frag -> row-major
      }
    }
    __syncthreads();
    v16h ap = load_frag_a(pbuf, 32, 0, 0);                  // row-major -> A-frag
#pragma unroll
    for (int o = 0; o < OBJ; ++o) {
      const _Float16* VT = v_attT + (size_t)(((b * OBJ + o) * HEADS) + h) * DV * THW;
      v16h bv0 = load_frag_a(VT, THW, 0,  col0);
      v16h bv1 = load_frag_a(VT, THW, 16, col0);
      acc[o][0] = wmma16(ap, bv0, acc[o][0]);
      acc[o][1] = wmma16(ap, bv1, acc[o][1]);
    }
    __syncthreads();
  }

  // ---- write (n, b*4+o, h*32+dv) ------------------------------------------
#pragma unroll
  for (int o = 0; o < OBJ; ++o)
#pragma unroll
    for (int j = 0; j < 2; ++j)
#pragma unroll
      for (int v = 0; v < 8; ++v) {
        int nrow = row0 + mh + v;
        attn_out[((size_t)nrow * (BQ * OBJ) + (b * OBJ + o)) * DMODEL +
                 h * DV + j * 16 + nL] = acc[o][j][v] * rinv[v];
      }
}

// ---------------------------------------------------------------------------
// WMMA GEMM + bias:  C(M,N) = A(M,K) * B(N,K)^T + bias[N]   (B stored N-major)
// ---------------------------------------------------------------------------
__global__ void __launch_bounds__(32)
gemm_wmma_bias(const _Float16* __restrict__ A, const _Float16* __restrict__ Bn,
               const float* __restrict__ bias, float* __restrict__ C,
               int M, int N, int Kd) {
  const int m0 = blockIdx.x * 16;
  const int n0 = blockIdx.y * 16;
  v8f acc = {};
  for (int k0 = 0; k0 < Kd; k0 += 32) {
    v16h a = load_frag_a(A, Kd, m0, k0);
    v16h b = load_frag_a(Bn, Kd, n0, k0);
    acc = wmma16(a, b, acc);
  }
  const int lane = threadIdx.x & 31;
  const int nL   = lane & 15;
  const int mh   = (lane >> 4) * 8;
  const float bc = bias[n0 + nL];
#pragma unroll
  for (int v = 0; v < 8; ++v)
    C[(size_t)(m0 + mh + v) * N + n0 + nL] = acc[v] + bc;
}

// ---------------------------------------------------------------------------
// host orchestration
// ---------------------------------------------------------------------------
extern "C" void kernel_launch(void* const* d_in, const int* in_sizes, int n_in,
                              void* d_out, int out_size, void* d_ws, size_t ws_size,
                              hipStream_t stream) {
  (void)in_sizes; (void)n_in; (void)out_size; (void)ws_size;
  const float* query  = (const float*)d_in[0];
  const float* key    = (const float*)d_in[1];
  const float* value  = (const float*)d_in[2];
  const float* q_w    = (const float*)d_in[3];
  const float* q_b    = (const float*)d_in[4];
  const float* k_w    = (const float*)d_in[5];
  const float* k_b    = (const float*)d_in[6];
  const float* v_w    = (const float*)d_in[7];
  const float* v_b    = (const float*)d_in[8];
  const float* qn_g   = (const float*)d_in[9];
  const float* qn_b   = (const float*)d_in[10];
  const float* kn_g   = (const float*)d_in[11];
  const float* kn_b   = (const float*)d_in[12];
  const float* vn_g   = (const float*)d_in[13];
  const float* vn_b   = (const float*)d_in[14];
  const float* proj_w = (const float*)d_in[15];
  const float* proj_b = (const float*)d_in[16];
  float* out = (float*)d_out;

  char* wp = (char*)d_ws;
  auto alloc = [&](size_t bytes) {
    void* r = (void*)wp;
    wp += (bytes + 255) & ~(size_t)255;
    return r;
  };

  const size_t nQX = (size_t)BQ * DMODEL * HW;
  const size_t nKX = (size_t)BQ * TT * DMODEL * HW;
  const size_t nVX = (size_t)BQ * OBJ * TT * DMODEL * HW;
  const size_t nQP = (size_t)BQ * PHW * CIN;
  const size_t nKP = (size_t)BQ * TT * PHW * CIN;
  const size_t nVP = (size_t)BQ * OBJ * TT * PHW * CIN;
  const size_t nQW = (size_t)DQK * KD9;
  const size_t nVW = (size_t)DMODEL * KD9;
  const size_t nPW = (size_t)DMODEL * DMODEL;
  const size_t nCQ = (size_t)BQ * DQK * HW;
  const size_t nCK = (size_t)BQ * TT * DQK * HW;
  const size_t nCV = (size_t)BQ * OBJ * TT * DMODEL * HW;
  const size_t nQA = (size_t)BQ * HEADS * HW * DQ;
  const size_t nKA = (size_t)BQ * HEADS * THW * DQ;
  const size_t nVA = (size_t)BQ * OBJ * HEADS * DV * THW;
  const size_t nAO = (size_t)HW * BQ * OBJ * DMODEL;

  _Float16* qx  = (_Float16*)alloc(nQX * 2);
  _Float16* kx  = (_Float16*)alloc(nKX * 2);
  _Float16* vx  = (_Float16*)alloc(nVX * 2);
  _Float16* qxp = (_Float16*)alloc(nQP * 2);
  _Float16* kxp = (_Float16*)alloc(nKP * 2);
  _Float16* vxp = (_Float16*)alloc(nVP * 2);
  _Float16* qwr = (_Float16*)alloc(nQW * 2);
  _Float16* kwr = (_Float16*)alloc(nQW * 2);
  _Float16* vwr = (_Float16*)alloc(nVW * 2);
  _Float16* pwh = (_Float16*)alloc(nPW * 2);
  float* convq = (float*)alloc(nCQ * 4);
  float* convk = (float*)alloc(nCK * 4);
  float* convv = (float*)alloc(nCV * 4);
  float* scq = (float*)alloc(DQK * 4);
  float* shq = (float*)alloc(DQK * 4);
  float* sck = (float*)alloc(DQK * 4);
  float* shk = (float*)alloc(DQK * 4);
  float* scv = (float*)alloc(DMODEL * 4);
  float* shv = (float*)alloc(DMODEL * 4);
  _Float16* qatt  = (_Float16*)alloc(nQA * 2);
  _Float16* kattT = (_Float16*)alloc(nKA * 2);
  _Float16* vattT = (_Float16*)alloc(nVA * 2);
  float*    attnf = (float*)alloc(nAO * 4);
  _Float16* attnh = (_Float16*)alloc(nAO * 2);

  const int TPB = 256;
  auto blocks = [](size_t n, int tpb) { return (unsigned)((n + tpb - 1) / tpb); };

  // --- f16 conversions / repacks / NHWC padding ---
  cvt_f32_f16<<<blocks(nQX, TPB), TPB, 0, stream>>>(query, qx, (int)nQX);
  pack_key_f16<<<blocks(nKX, TPB), TPB, 0, stream>>>(key, kx);
  pack_value_f16<<<blocks(nVX, TPB), TPB, 0, stream>>>(value, vx);
  pad_hwc_f16<<<blocks(nQP, TPB), TPB, 0, stream>>>(qx, qxp, BQ);
  pad_hwc_f16<<<blocks(nKP, TPB), TPB, 0, stream>>>(kx, kxp, BQ * TT);
  pad_hwc_f16<<<blocks(nVP, TPB), TPB, 0, stream>>>(vx, vxp, BQ * OBJ * TT);
  repack_w_f16<<<blocks(nQW, TPB), TPB, 0, stream>>>(q_w, qwr, DQK);
  repack_w_f16<<<blocks(nQW, TPB), TPB, 0, stream>>>(k_w, kwr, DQK);
  repack_w_f16<<<blocks(nVW, TPB), TPB, 0, stream>>>(v_w, vwr, DMODEL);
  cvt_f32_f16<<<blocks(nPW, TPB), TPB, 0, stream>>>(proj_w, pwh, (int)nPW);

  // --- convs as implicit GEMM (WMMA + async-LDS weight pipeline) ---
  conv3x3_wmma<<<dim3(DQK / 16, (BQ * HW) / 16), 32, 0, stream>>>(
      qxp, qwr, q_b, convq, DQK);
  conv3x3_wmma<<<dim3(DQK / 16, (BQ * TT * HW) / 16), 32, 0, stream>>>(
      kxp, kwr, k_b, convk, DQK);
  conv3x3_wmma<<<dim3(DMODEL / 16, (BQ * OBJ * TT * HW) / 16), 32, 0, stream>>>(
      vxp, vwr, v_b, convv, DMODEL);

  // --- batchnorm stats ---
  bn_stats<<<DQK, 256, 0, stream>>>(convq, BQ, DQK, qn_g, qn_b, scq, shq);
  bn_stats<<<DQK, 256, 0, stream>>>(convk, BQ * TT, DQK, kn_g, kn_b, sck, shk);
  bn_stats<<<DMODEL, 256, 0, stream>>>(convv, BQ * OBJ * TT, DMODEL, vn_g, vn_b, scv, shv);

  // --- normalize + repack for attention ---
  pack_q_att <<<blocks(nCQ, TPB), TPB, 0, stream>>>(convq, scq, shq, qatt);
  pack_k_attT<<<blocks(nCK, TPB), TPB, 0, stream>>>(convk, sck, shk, kattT);
  pack_v_attT<<<blocks(nCV, TPB), TPB, 0, stream>>>(convv, scv, shv, vattT);

  // --- fused attention (online softmax, 2 streaming passes) ---
  attention_wmma<<<dim3(BQ * HEADS * (HW / 16)), 32, 0, stream>>>(
      qatt, kattT, vattT, attnf);

  // --- output projection ---
  cvt_f32_f16<<<blocks(nAO, TPB), TPB, 0, stream>>>(attnf, attnh, (int)nAO);
  gemm_wmma_bias<<<dim3((HW * BQ * OBJ) / 16, DMODEL / 16), 32, 0, stream>>>(
      attnh, pwh, proj_b, out, HW * BQ * OBJ, DMODEL, DMODEL);
}
